// TSTEncoderLayer_2379411882046
// MI455X (gfx1250) — compile-verified, matrix operand order checked by reading, not run on hardware
//
#include <hip/hip_runtime.h>
#include <stdint.h>

typedef __bf16 bf16;
typedef __attribute__((ext_vector_type(4)))  __bf16 v4bf;
typedef __attribute__((ext_vector_type(8)))  __bf16 v8bf;
typedef __attribute__((ext_vector_type(16))) __bf16 v16bf;
typedef __attribute__((ext_vector_type(8)))  float  v8f;

static constexpr int BB = 8, SSEQ = 1024, DD = 512, HH = 8, DFF = 2048, DKK = 64;
static constexpr int ROWS = BB * SSEQ;          // 8192 tokens
static constexpr float EPSBN = 1e-5f;

// A-tile source modes
static constexpr int A_F32 = 0;      // fp32 global -> regs -> bf16 LDS
static constexpr int A_BF16 = 1;     // bf16 global -> LDS via async copy
static constexpr int A_SOFTMAX = 2;  // fp32 scores -> exp(s-max)*inv -> bf16 LDS
// B-tile source modes
static constexpr int B_N_F32 = 0;    // B is KxN fp32 (weight); 4x4 reg-transpose into LDS
static constexpr int B_N_BF16 = 1;   // B is KxN bf16; 4x4 reg-transpose into LDS
static constexpr int B_T_BF16 = 2;   // B is NxK bf16; async copy (C = A * B^T)
// Epilogues
static constexpr int EPI_BF16 = 0;       // store bf16
static constexpr int EPI_F32 = 1;        // store fp32
static constexpr int EPI_BIAS_F32 = 2;   // + bias[col], store fp32
static constexpr int EPI_GELU_BF16 = 3;  // gelu(acc + bias[col]), store bf16
static constexpr int EPI_SCORES = 4;     // acc*scale + prev[same idx], store fp32

__device__ __forceinline__ v16bf load_frag(const bf16* tile, int rowBase, int lane) {
  // tile is row-major [rows][32] bf16. 16-bit A/B fragment layout:
  // lanes 0-15:  row = rowBase+lane,      K chunks {0..7, 16..23}
  // lanes 16-31: row = rowBase+lane-16,   K chunks {8..15, 24..31}
  int half = (lane >> 4) & 1;
  const bf16* p = tile + (rowBase + (lane & 15)) * 32;
  v8bf lo = *(const v8bf*)(p + half * 8);
  v8bf hi = *(const v8bf*)(p + 16 + half * 8);
  return __builtin_shufflevector(lo, hi, 0,1,2,3,4,5,6,7,8,9,10,11,12,13,14,15);
}

template<int BM, int BN, int AM, int BMODE, int EPI>
__global__ __launch_bounds__(256) void gemm_wmma(
    const void* __restrict__ A, long long aOff, long long aOuter, long long aInner, int lda,
    const void* __restrict__ B, long long bOff, long long bOuter, long long bInner, int ldb,
    void* __restrict__ C,       long long cOff, long long cOuter, long long cInner, int ldc,
    const float* __restrict__ bias,       // bias[N] or prev (for EPI_SCORES)
    const float* __restrict__ rowMax, const float* __restrict__ rowInv, int statStride,
    int K, int zDiv, float scale)
{
  static_assert((BM / 32) * (BN / 32) == 8, "8 waves per block");
  __shared__ bf16 As[2][BM][32];
  __shared__ bf16 Bs[2][BN][32];

  const int tid = threadIdx.x;
  const int lane = tid & 31;
  const int w = tid >> 5;
  constexpr int NWM = BM / 32;
  const int wm = (w % NWM) * 32;
  const int wn = (w / NWM) * 32;

  const int z = blockIdx.z;
  const int zo = z / zDiv, zi = z % zDiv;
  const long long aBase = aOff + (long long)zo * aOuter + (long long)zi * aInner;
  const long long bBase = bOff + (long long)zo * bOuter + (long long)zi * bInner;
  const long long cBase = cOff + (long long)zo * cOuter + (long long)zi * cInner;
  const long long statBase = (long long)z * statStride;

  const int m0 = blockIdx.y * BM;
  const int n0 = blockIdx.x * BN;

  constexpr bool ASYNC_A = (AM == A_BF16);
  constexpr bool ASYNC_B = (BMODE == B_T_BF16);
  constexpr bool ANY_ASYNC = ASYNC_A || ASYNC_B;
  constexpr bool BTRANS   = (BMODE == B_N_F32 || BMODE == B_N_BF16);
  constexpr int NA4 = ASYNC_A ? 0 : (BM * 8) / 256;  // float4 groups per thread (A)

  float4 aReg[NA4 ? NA4 : 1];
  float  mxR [NA4 ? NA4 : 1];
  float  invR[NA4 ? NA4 : 1];
  float4 bReg4 [BTRANS ? 4 : 1];   // 4 k-rows x 4 n-cols fp32 sub-block
  v4bf   bReg4h[BTRANS ? 4 : 1];   // same, bf16 source

  // B 4x4 sub-block ownership: 2*BN sub-blocks per 32-deep K tile
  const int bkq = tid / (BN / 4);        // k group 0..7
  const int bnq = tid % (BN / 4);        // n group
  const bool bAct = BTRANS && (tid < 2 * BN);

  // ---- async bf16 tile copy: global -> LDS, 16B per lane per issue (ASYNCcnt) ----
  auto issue_async = [&](int kt, int buf) {
    if (ASYNC_A) {
      uint32_t base = (uint32_t)(uintptr_t)(&As[buf][0][0]);
      #pragma unroll
      for (int j = 0; j < (ASYNC_A ? BM / 64 : 1); j++) {
        int g = tid + j * 256;
        int row = g >> 2, c8 = (g & 3) * 8;
        unsigned long long gp =
            (unsigned long long)((const bf16*)A + aBase + (long long)(m0 + row) * lda + kt + c8);
        uint32_t lp = base + (uint32_t)g * 16;
        asm volatile("global_load_async_to_lds_b128 %0, %1, off"
                     :: "v"(lp), "v"(gp) : "memory");
      }
    }
    if (ASYNC_B) {
      uint32_t base = (uint32_t)(uintptr_t)(&Bs[buf][0][0]);
      #pragma unroll
      for (int j = 0; j < (ASYNC_B ? BN / 64 : 1); j++) {
        int g = tid + j * 256;
        int row = g >> 2, c8 = (g & 3) * 8;
        unsigned long long gp =
            (unsigned long long)((const bf16*)B + bBase + (long long)(n0 + row) * ldb + kt + c8);
        uint32_t lp = base + (uint32_t)g * 16;
        asm volatile("global_load_async_to_lds_b128 %0, %1, off"
                     :: "v"(lp), "v"(gp) : "memory");
      }
    }
  };

  // ---- fp32 / transposing paths: global -> registers ----
  auto load_regs = [&](int kt) {
    if (NA4) {
      #pragma unroll
      for (int j = 0; j < (NA4 ? NA4 : 1); j++) {
        int g = tid + j * 256;
        int row = g >> 3, col = (g & 7) << 2;
        aReg[j] = *(const float4*)((const float*)A + aBase + (long long)(m0 + row) * lda + kt + col);
        if (AM == A_SOFTMAX) {
          mxR[j]  = rowMax[statBase + m0 + row];
          invR[j] = rowInv[statBase + m0 + row];
        }
      }
    }
    if (BTRANS) {
      if (bAct) {
        #pragma unroll
        for (int i = 0; i < 4; i++) {   // 4 consecutive k-rows, 4 n-cols each
          long long off = bBase + (long long)(kt + bkq * 4 + i) * ldb + n0 + bnq * 4;
          if (BMODE == B_N_F32) bReg4[i]  = *(const float4*)((const float*)B + off);
          else                  bReg4h[i] = *(const v4bf*)((const bf16*)B + off);
        }
      }
    }
  };

  // ---- registers -> bf16 LDS (A convert, B 4x4 register transpose) ----
  auto store_regs = [&](int buf) {
    if (NA4) {
      #pragma unroll
      for (int j = 0; j < (NA4 ? NA4 : 1); j++) {
        int g = tid + j * 256;
        int row = g >> 3, col = (g & 7) << 2;
        float4 t = aReg[j];
        if (AM == A_SOFTMAX) {
          float mx = mxR[j], inv = invR[j];
          t.x = __expf(t.x - mx) * inv; t.y = __expf(t.y - mx) * inv;
          t.z = __expf(t.z - mx) * inv; t.w = __expf(t.w - mx) * inv;
        }
        v4bf p;
        p[0] = (bf16)t.x; p[1] = (bf16)t.y; p[2] = (bf16)t.z; p[3] = (bf16)t.w;
        *(v4bf*)&As[buf][row][col] = p;
      }
    }
    if (BTRANS) {
      if (bAct) {
        int k = bkq * 4, n = bnq * 4;
        #pragma unroll
        for (int i = 0; i < 4; i++) {          // output row n+i, k..k+3 contiguous
          v4bf t;
          if (BMODE == B_N_F32) {
            t[0] = (bf16)(((const float*)&bReg4[0])[i]);
            t[1] = (bf16)(((const float*)&bReg4[1])[i]);
            t[2] = (bf16)(((const float*)&bReg4[2])[i]);
            t[3] = (bf16)(((const float*)&bReg4[3])[i]);
          } else {
            t[0] = bReg4h[0][i]; t[1] = bReg4h[1][i];
            t[2] = bReg4h[2][i]; t[3] = bReg4h[3][i];
          }
          *(v4bf*)&Bs[buf][n + i][k] = t;      // 8-byte LDS store
        }
      }
    }
  };

  v8f acc[2][2] = {};

  // prologue: tile 0 -> buffer 0
  load_regs(0);
  issue_async(0, 0);
  store_regs(0);
  if (ANY_ASYNC) asm volatile("s_wait_asynccnt 0" ::: "memory");
  __syncthreads();

  int cur = 0;
  for (int kt = 0; kt < K; kt += 32) {
    const bool hasNext = (kt + 32) < K;
    if (hasNext) {             // start fetching next tile before compute
      load_regs(kt + 32);
      issue_async(kt + 32, cur ^ 1);
    }
    const bf16* At = &As[cur][0][0];
    const bf16* Bt = &Bs[cur][0][0];
    v16bf a0 = load_frag(At, wm,      lane);
    v16bf a1 = load_frag(At, wm + 16, lane);
    v16bf b0 = load_frag(Bt, wn,      lane);
    v16bf b1 = load_frag(Bt, wn + 16, lane);
    acc[0][0] = __builtin_amdgcn_wmma_f32_16x16x32_bf16(false, a0, false, b0, (short)0, acc[0][0], false, false);
    acc[0][1] = __builtin_amdgcn_wmma_f32_16x16x32_bf16(false, a0, false, b1, (short)0, acc[0][1], false, false);
    acc[1][0] = __builtin_amdgcn_wmma_f32_16x16x32_bf16(false, a1, false, b0, (short)0, acc[1][0], false, false);
    acc[1][1] = __builtin_amdgcn_wmma_f32_16x16x32_bf16(false, a1, false, b1, (short)0, acc[1][1], false, false);
    if (hasNext) {
      store_regs(cur ^ 1);
      if (ANY_ASYNC) asm volatile("s_wait_asynccnt 0" ::: "memory");
      __syncthreads();
      cur ^= 1;
    }
  }

  // ---- epilogue: C/D 16x16 f32 layout: VGPR v -> M = v + 8*half, N = lane&15 ----
  const int half = lane >> 4;
  #pragma unroll
  for (int mi = 0; mi < 2; mi++) {
    #pragma unroll
    for (int ni = 0; ni < 2; ni++) {
      int gc = n0 + wn + ni * 16 + (lane & 15);
      #pragma unroll
      for (int v = 0; v < 8; v++) {
        int gr = m0 + wm + mi * 16 + half * 8 + v;
        long long ci = cBase + (long long)gr * ldc + gc;
        float val = acc[mi][ni][v];
        if (EPI == EPI_SCORES) {
          ((float*)C)[ci] = val * scale + bias[ci];
        } else if (EPI == EPI_F32) {
          ((float*)C)[ci] = val;
        } else if (EPI == EPI_BF16) {
          ((bf16*)C)[ci] = (bf16)val;
        } else if (EPI == EPI_BIAS_F32) {
          ((float*)C)[ci] = val + bias[gc];
        } else { // EPI_GELU_BF16 (exact gelu)
          float g = val + bias[gc];
          g = 0.5f * g * (1.0f + erff(g * 0.70710678118654752f));
          ((bf16*)C)[ci] = (bf16)g;
        }
      }
    }
  }
}

// ---- row softmax statistics: one wave per row of 1024 scores ----
__global__ __launch_bounds__(256) void softmax_stats_kernel(
    const float* __restrict__ scores, float* __restrict__ rowMax, float* __restrict__ rowInv)
{
  int row = blockIdx.x * 8 + (threadIdx.x >> 5);
  int lane = threadIdx.x & 31;
  const float* p = scores + (long long)row * SSEQ;
  float mx = -3.4e38f;
  #pragma unroll 4
  for (int j = lane; j < SSEQ; j += 32) mx = fmaxf(mx, p[j]);
  #pragma unroll
  for (int o = 16; o > 0; o >>= 1) mx = fmaxf(mx, __shfl_xor(mx, o, 32));
  float s = 0.f;
  #pragma unroll 4
  for (int j = lane; j < SSEQ; j += 32) s += __expf(p[j] - mx);
  #pragma unroll
  for (int o = 16; o > 0; o >>= 1) s += __shfl_xor(s, o, 32);
  if (lane == 0) { rowMax[row] = mx; rowInv[row] = 1.0f / s; }
}

__global__ void zero_kernel(float* p, int n) {
  int i = blockIdx.x * blockDim.x + threadIdx.x;
  if (i < n) p[i] = 0.0f;
}

// ---- per-channel sum / sumsq of (a + b), channels = D = 512 ----
__global__ __launch_bounds__(512) void bn_reduce_kernel(
    const float* __restrict__ a, const float* __restrict__ b,
    float* __restrict__ accSum, float* __restrict__ accSq)
{
  int c = threadIdx.x;               // 0..511 == channel
  int r0 = blockIdx.x * 128;
  float s = 0.f, q = 0.f;
  #pragma unroll 4
  for (int r = 0; r < 128; r++) {
    long long idx = (long long)(r0 + r) * DD + c;
    float t = a[idx] + b[idx];
    s += t; q += t * t;
  }
  atomicAdd(&accSum[c], s);
  atomicAdd(&accSq[c], q);
}

__global__ __launch_bounds__(256) void bn_norm_kernel(
    const float* __restrict__ a, const float* __restrict__ b,
    const float* __restrict__ gamma, const float* __restrict__ beta,
    const float* __restrict__ accSum, const float* __restrict__ accSq,
    float* __restrict__ outF, bf16* __restrict__ outB)
{
  long long i = (long long)blockIdx.x * blockDim.x + threadIdx.x;
  int c = (int)(i & (DD - 1));
  const float invN = 1.0f / (float)ROWS;
  float mean = accSum[c] * invN;
  float var  = accSq[c] * invN - mean * mean;
  float t = a[i] + b[i];
  float y = (t - mean) * rsqrtf(var + EPSBN) * gamma[c] + beta[c];
  outF[i] = y;
  if (outB) outB[i] = (bf16)y;
}

extern "C" void kernel_launch(void* const* d_in, const int* in_sizes, int n_in,
                              void* d_out, int out_size, void* d_ws, size_t ws_size,
                              hipStream_t stream) {
  const float* src   = (const float*)d_in[0];
  const float* prev  = (const float*)d_in[1];
  const float* Wq    = (const float*)d_in[2];
  const float* Wk    = (const float*)d_in[3];
  const float* Wv    = (const float*)d_in[4];
  const float* Wo    = (const float*)d_in[5];
  const float* bn1w  = (const float*)d_in[6];
  const float* bn1b  = (const float*)d_in[7];
  const float* fw1   = (const float*)d_in[8];
  const float* fb1   = (const float*)d_in[9];
  const float* fw2   = (const float*)d_in[10];
  const float* fb2   = (const float*)d_in[11];
  const float* bn2w  = (const float*)d_in[12];
  const float* bn2b  = (const float*)d_in[13];

  float* outMain   = (float*)d_out;                       // (B,S,D)
  float* outScores = outMain + (long long)ROWS * DD;      // (B,H,S,S)

  // ---- workspace carve-up ----
  char* w = (char*)d_ws;
  auto carve = [&](size_t bytes) { void* p = (void*)w; w += (bytes + 255) & ~(size_t)255; return p; };
  bf16*  qb      = (bf16*)carve((size_t)ROWS * DD * 2);
  bf16*  kb      = (bf16*)carve((size_t)ROWS * DD * 2);
  bf16*  vb      = (bf16*)carve((size_t)ROWS * DD * 2);
  bf16*  ctxb    = (bf16*)carve((size_t)ROWS * DD * 2);
  bf16*  xb      = (bf16*)carve((size_t)ROWS * DD * 2);
  bf16*  hb      = (bf16*)carve((size_t)ROWS * DFF * 2);
  float* attnOut = (float*)carve((size_t)ROWS * DD * 4);
  float* xf      = (float*)carve((size_t)ROWS * DD * 4);
  float* ff2     = (float*)carve((size_t)ROWS * DD * 4);
  float* rowMax  = (float*)carve((size_t)BB * HH * SSEQ * 4);
  float* rowInv  = (float*)carve((size_t)BB * HH * SSEQ * 4);
  float* acc     = (float*)carve(4 * DD * 4);   // [sum1|sq1|sum2|sq2]

  const long long SD  = (long long)SSEQ * DD;         // 524288
  const long long SS2 = (long long)SSEQ * SSEQ;       // 1048576
  const float scale = 0.125f;                         // DK^-0.5

  // 0) zero BN accumulators (re-zeroed every call for determinism)
  zero_kernel<<<dim3((4 * DD + 255) / 256), 256, 0, stream>>>(acc, 4 * DD);

  // 1) Q,K,V projections: (8192x512) @ (512x512) -> bf16
  dim3 gProj(DD / 128, ROWS / 64, 1);
  gemm_wmma<64,128,A_F32,B_N_F32,EPI_BF16><<<gProj, 256, 0, stream>>>(
      src, 0,0,0, DD,  Wq, 0,0,0, DD,  qb, 0,0,0, DD,
      nullptr, nullptr, nullptr, 0, DD, 1, 0.f);
  gemm_wmma<64,128,A_F32,B_N_F32,EPI_BF16><<<gProj, 256, 0, stream>>>(
      src, 0,0,0, DD,  Wk, 0,0,0, DD,  kb, 0,0,0, DD,
      nullptr, nullptr, nullptr, 0, DD, 1, 0.f);
  gemm_wmma<64,128,A_F32,B_N_F32,EPI_BF16><<<gProj, 256, 0, stream>>>(
      src, 0,0,0, DD,  Wv, 0,0,0, DD,  vb, 0,0,0, DD,
      nullptr, nullptr, nullptr, 0, DD, 1, 0.f);

  // 2) scores = Q @ K^T * scale + prev, batched over (b,h); written to output
  //    (both A and B tiles travel via global_load_async_to_lds_b128)
  dim3 gScores(SSEQ / 128, SSEQ / 64, BB * HH);
  gemm_wmma<64,128,A_BF16,B_T_BF16,EPI_SCORES><<<gScores, 256, 0, stream>>>(
      qb, 0, SD, DKK, DD,
      kb, 0, SD, DKK, DD,
      outScores, 0, (long long)HH * SS2, SS2, SSEQ,
      prev, nullptr, nullptr, 0, DKK, HH, scale);

  // 3) row softmax statistics (max, 1/sum)
  softmax_stats_kernel<<<dim3(BB * HH * SSEQ / 8), 256, 0, stream>>>(outScores, rowMax, rowInv);

  // 4) ctx = softmax(scores) @ V, batched over (b,h): M=1024,N=64,K=1024
  dim3 gCtx(DKK / 64, SSEQ / 128, BB * HH);
  gemm_wmma<128,64,A_SOFTMAX,B_N_BF16,EPI_BF16><<<gCtx, 256, 0, stream>>>(
      outScores, 0, (long long)HH * SS2, SS2, SSEQ,
      vb, 0, SD, DKK, DD,
      ctxb, 0, SD, DKK, DD,
      nullptr, rowMax, rowInv, SSEQ, SSEQ, HH, 0.f);

  // 5) attn_out = ctx @ Wo -> fp32  (A tile async-to-LDS)
  gemm_wmma<64,128,A_BF16,B_N_F32,EPI_F32><<<gProj, 256, 0, stream>>>(
      ctxb, 0,0,0, DD,  Wo, 0,0,0, DD,  attnOut, 0,0,0, DD,
      nullptr, nullptr, nullptr, 0, DD, 1, 0.f);

  // 6) BN1 over (src + attn_out): reduce, then normalize -> x (fp32 + bf16)
  bn_reduce_kernel<<<dim3(ROWS / 128), 512, 0, stream>>>(src, attnOut, acc, acc + DD);
  bn_norm_kernel<<<dim3((unsigned)((long long)ROWS * DD / 256)), 256, 0, stream>>>(
      src, attnOut, bn1w, bn1b, acc, acc + DD, xf, xb);

  // 7) FFN1: h = gelu(x @ W1 + b1) -> bf16 (8192x512 @ 512x2048)
  dim3 gFF1(DFF / 128, ROWS / 64, 1);
  gemm_wmma<64,128,A_BF16,B_N_F32,EPI_GELU_BF16><<<gFF1, 256, 0, stream>>>(
      xb, 0,0,0, DD,  fw1, 0,0,0, DFF,  hb, 0,0,0, DFF,
      fb1, nullptr, nullptr, 0, DD, 1, 0.f);

  // 8) FFN2: ff2 = h @ W2 + b2 -> fp32 (8192x2048 @ 2048x512)
  gemm_wmma<64,128,A_BF16,B_N_F32,EPI_BIAS_F32><<<gProj, 256, 0, stream>>>(
      hb, 0,0,0, DFF,  fw2, 0,0,0, DD,  ff2, 0,0,0, DD,
      fb2, nullptr, nullptr, 0, DFF, 1, 0.f);

  // 9) BN2 over (x + ff2) -> final output
  bn_reduce_kernel<<<dim3(ROWS / 128), 512, 0, stream>>>(xf, ff2, acc + 2 * DD, acc + 3 * DD);
  bn_norm_kernel<<<dim3((unsigned)((long long)ROWS * DD / 256)), 256, 0, stream>>>(
      xf, ff2, bn2w, bn2b, acc + 2 * DD, acc + 3 * DD, outMain, nullptr);

  (void)in_sizes; (void)n_in; (void)out_size; (void)ws_size;
}
